// RWRLayer_74122545594652
// MI455X (gfx1250) — matrix operand order repaired and checked by reading, not compiled, not run on hardware
//
#include <hip/hip_runtime.h>
#include <cstdint>

// ---------------- problem constants ----------------
constexpr int NN    = 8192;   // nodes
constexpr int INF_  = 512;    // input features
constexpr int OF    = 64;     // output features
constexpr int SPLITK = 8;
constexpr int KSLICE = NN / SPLITK;   // 1024
constexpr float NSLOPE = 0.2f;

typedef __attribute__((ext_vector_type(16))) __bf16        v16bf;
typedef __attribute__((ext_vector_type(8)))  float         v8f;
typedef __attribute__((ext_vector_type(8)))  unsigned int  v8u;

// pack two f32 -> bf16x2 dword: +0x8000 round bias, one v_perm_b32
__device__ __forceinline__ unsigned int pk_bf16(float a, float b) {
    unsigned ua = __builtin_bit_cast(unsigned, a) + 0x8000u;
    unsigned ub = __builtin_bit_cast(unsigned, b) + 0x8000u;
    return __builtin_amdgcn_perm(ub, ua, 0x07060302u);  // [ub.hi16, ua.hi16]
}

__device__ __forceinline__ float lrelu(float t) { return fmaxf(t, NSLOPE * t); }

// ------------------------------------------------------------------
// Kernel 1: pack W (512x64 f32) into WMMA B-operand bf16 layout.
// dword index = ((kt*4 + nt)*32 + lane)*8 + v
//   lane<16: N = lane,    K = kt*32 + 2v{,+1} (+8 if v>=4)
//   lane>=16:N = lane-16, K = kt*32 + 8 + 2v{,+1} (+8 if v>=4)
// ------------------------------------------------------------------
__global__ void __launch_bounds__(256) prep_w_kernel(const float* __restrict__ W,
                                                     unsigned* __restrict__ Wpk) {
    int idx  = blockIdx.x * 256 + threadIdx.x;     // 16384 total
    int v    = idx & 7;
    int lane = (idx >> 3) & 31;
    int nt   = (idx >> 8) & 3;
    int kt   = idx >> 10;
    int q    = lane & 15;
    int koff = (lane >> 4) * 8;
    int k    = kt * 32 + koff + 2 * v + (v & 4) * 2;
    int n    = nt * 16 + q;
    Wpk[idx] = pk_bf16(W[k * OF + n], W[(k + 1) * OF + n]);
}

// ------------------------------------------------------------------
// Kernel 2: h = x @ W via v_wmma_f32_16x16x32_bf16.
// Emits e1 = h@a1, e2 = h@a2, and h repacked as bf16 B-operand tiles.
// One wave per 16-row M-tile; 8 waves per block.
// ------------------------------------------------------------------
__global__ void __launch_bounds__(256) gemm1_kernel(const float* __restrict__ x,
                                                    const unsigned* __restrict__ Wpk,
                                                    const float* __restrict__ avec,
                                                    unsigned* __restrict__ hB,
                                                    float* __restrict__ e1,
                                                    float* __restrict__ e2) {
    const int l     = threadIdx.x & 31;
    const int wave  = threadIdx.x >> 5;
    const int mtile = blockIdx.x * 8 + wave;
    const int m0    = mtile * 16;
    const int q     = l & 15;
    const int half  = l >> 4;

    v8f acc0 = {}, acc1 = {}, acc2 = {}, acc3 = {};

    const float* xr = x + (size_t)(m0 + q) * INF_ + half * 8;

#pragma unroll
    for (int kt = 0; kt < INF_ / 32; ++kt) {
        const float* xp = xr + kt * 32;
        float4 xa = *(const float4*)(xp);
        float4 xb = *(const float4*)(xp + 4);
        float4 xc = *(const float4*)(xp + 16);
        float4 xd = *(const float4*)(xp + 20);
        v8u au;
        au[0] = pk_bf16(xa.x, xa.y); au[1] = pk_bf16(xa.z, xa.w);
        au[2] = pk_bf16(xb.x, xb.y); au[3] = pk_bf16(xb.z, xb.w);
        au[4] = pk_bf16(xc.x, xc.y); au[5] = pk_bf16(xc.z, xc.w);
        au[6] = pk_bf16(xd.x, xd.y); au[7] = pk_bf16(xd.z, xd.w);
        v16bf A = __builtin_bit_cast(v16bf, au);

        const v8u* bp = (const v8u*)Wpk + ((size_t)kt * 128 + l);
        v16bf B0 = __builtin_bit_cast(v16bf, bp[0]);
        v16bf B1 = __builtin_bit_cast(v16bf, bp[32]);
        v16bf B2 = __builtin_bit_cast(v16bf, bp[64]);
        v16bf B3 = __builtin_bit_cast(v16bf, bp[96]);

        acc0 = __builtin_amdgcn_wmma_f32_16x16x32_bf16(false, A, false, B0, (short)0, acc0, false, false);
        acc1 = __builtin_amdgcn_wmma_f32_16x16x32_bf16(false, A, false, B1, (short)0, acc1, false, false);
        acc2 = __builtin_amdgcn_wmma_f32_16x16x32_bf16(false, A, false, B2, (short)0, acc2, false, false);
        acc3 = __builtin_amdgcn_wmma_f32_16x16x32_bf16(false, A, false, B3, (short)0, acc3, false, false);
    }

    // ---- e1/e2: rows of this tile dotted with a1/a2, reduced across lanes ----
    float a10 = avec[q], a11 = avec[16 + q], a12 = avec[32 + q], a13 = avec[48 + q];
    float a20 = avec[64 + q], a21 = avec[80 + q], a22 = avec[96 + q], a23 = avec[112 + q];

    float p1[8], p2[8];
#pragma unroll
    for (int r = 0; r < 8; ++r) {
        p1[r] = acc0[r] * a10 + acc1[r] * a11 + acc2[r] * a12 + acc3[r] * a13;
        p2[r] = acc0[r] * a20 + acc1[r] * a21 + acc2[r] * a22 + acc3[r] * a23;
#pragma unroll
        for (int off = 1; off < 16; off <<= 1) {
            p1[r] += __shfl_xor(p1[r], off, 32);
            p2[r] += __shfl_xor(p2[r], off, 32);
        }
    }
    if (q == 0) {
        int base = m0 + 8 * half;
#pragma unroll
        for (int r = 0; r < 8; ++r) { e1[base + r] = p1[r]; e2[base + r] = p2[r]; }
    }

    // ---- repack h tile into bf16 B-operand layout for GEMM2 ----
    const int ktg = m0 >> 5;
    const int ll  = q + half * 16;   // target lane in B layout
    uint4 s0 = { pk_bf16(acc0[0], acc0[1]), pk_bf16(acc0[2], acc0[3]),
                 pk_bf16(acc0[4], acc0[5]), pk_bf16(acc0[6], acc0[7]) };
    uint4 s1 = { pk_bf16(acc1[0], acc1[1]), pk_bf16(acc1[2], acc1[3]),
                 pk_bf16(acc1[4], acc1[5]), pk_bf16(acc1[6], acc1[7]) };
    uint4 s2 = { pk_bf16(acc2[0], acc2[1]), pk_bf16(acc2[2], acc2[3]),
                 pk_bf16(acc2[4], acc2[5]), pk_bf16(acc2[6], acc2[7]) };
    uint4 s3 = { pk_bf16(acc3[0], acc3[1]), pk_bf16(acc3[2], acc3[3]),
                 pk_bf16(acc3[4], acc3[5]), pk_bf16(acc3[6], acc3[7]) };
    unsigned* op = hB + ((size_t)(ktg * 4) * 32 + ll) * 8 + ((m0 & 16) ? 4 : 0);
    *(uint4*)(op + 0 * 256) = s0;   // nt stride = 32 lanes * 8 dwords
    *(uint4*)(op + 1 * 256) = s1;
    *(uint4*)(op + 2 * 256) = s2;
    *(uint4*)(op + 3 * 256) = s3;
}

// ------------------------------------------------------------------
// Kernel 3: e2 prep: global max of e2 (softmax shift bound; lrelu is
// monotone) + factored exponentials g1 = exp(e2), g2 = exp(0.2*e2).
// ------------------------------------------------------------------
__global__ void __launch_bounds__(256) e2prep_kernel(const float* __restrict__ e2,
                                                     float* __restrict__ gmax,
                                                     float* __restrict__ g1,
                                                     float* __restrict__ g2) {
    __shared__ float sm[256];
    float m = -3.0e38f;
    for (int i = threadIdx.x; i < NN; i += 256) {
        float v = e2[i];
        m = fmaxf(m, v);
        g1[i] = __expf(v);
        g2[i] = __expf(NSLOPE * v);
    }
    sm[threadIdx.x] = m;
    __syncthreads();
    for (int s = 128; s > 0; s >>= 1) {
        if ((int)threadIdx.x < s) sm[threadIdx.x] = fmaxf(sm[threadIdx.x], sm[threadIdx.x + s]);
        __syncthreads();
    }
    if (threadIdx.x == 0) gmax[0] = sm[0];
}

// ------------------------------------------------------------------
// Kernel 4: fused masked-softmax numerator/denominator, split-K.
// grid = (512 M-tiles, SPLITK), one wave per block.
// Weight: exp(lrelu(e1+e2)-m) = max(c1*g1_j, c2*g2_j)  (no exp in loop)
// Row sums via a 5th WMMA against an all-ones bf16 B tile.
// ------------------------------------------------------------------
__global__ void __launch_bounds__(32) gemm2_kernel(const int* __restrict__ adj,
                                                   const unsigned* __restrict__ hB,
                                                   const float* __restrict__ e1,
                                                   const float* __restrict__ g1,
                                                   const float* __restrict__ g2,
                                                   const float* __restrict__ gmaxp,
                                                   float* __restrict__ pnum,
                                                   float* __restrict__ ps) {
    __shared__ float sg1[KSLICE];
    __shared__ float sg2[KSLICE];
    const int l    = threadIdx.x;
    const int q    = l & 15;
    const int half = l >> 4;
    const int m0   = blockIdx.x * 16;
    const int y    = blockIdx.y;
    const int k0   = y * KSLICE;

    // stage factored-exp slices in LDS (broadcast reads later)
#pragma unroll
    for (int t = l; t < KSLICE / 4; t += 32) {
        ((float4*)sg1)[t] = ((const float4*)(g1 + k0))[t];
        ((float4*)sg2)[t] = ((const float4*)(g2 + k0))[t];
    }
    __syncthreads();

    const int   i    = m0 + q;                  // A-layout row for this lane
    const float e1i  = e1[i];
    const float t0   = e1i + gmaxp[0];
    const float mrow = fmaxf(t0, NSLOPE * t0);  // valid softmax shift
    const float c1   = __expf(e1i - mrow);
    const float c2   = __expf(NSLOPE * e1i - mrow);

    const int* arow = adj + (size_t)i * NN + k0 + half * 8;

    // all-ones bf16 B tile, in registers
    v8u ones_u = { 0x3F803F80u, 0x3F803F80u, 0x3F803F80u, 0x3F803F80u,
                   0x3F803F80u, 0x3F803F80u, 0x3F803F80u, 0x3F803F80u };
    v16bf Bones = __builtin_bit_cast(v16bf, ones_u);

    v8f acc0 = {}, acc1 = {}, acc2 = {}, acc3 = {}, accS = {};

#pragma unroll 2
    for (int kt = 0; kt < KSLICE / 32; ++kt) {
        const int jb = kt * 32;
        int4 A0 = *(const int4*)(arow + jb);
        int4 A1 = *(const int4*)(arow + jb + 4);
        int4 A2 = *(const int4*)(arow + jb + 16);
        int4 A3 = *(const int4*)(arow + jb + 20);
        __builtin_prefetch(arow + jb + 32, 0, 0);   // next adj tile

        const float* f1p = sg1 + jb + half * 8;
        float4 F0 = *(const float4*)(f1p);
        float4 F1 = *(const float4*)(f1p + 4);
        float4 F2 = *(const float4*)(f1p + 16);
        float4 F3 = *(const float4*)(f1p + 20);
        const float* f2p = sg2 + jb + half * 8;
        float4 G0 = *(const float4*)(f2p);
        float4 G1 = *(const float4*)(f2p + 4);
        float4 G2 = *(const float4*)(f2p + 16);
        float4 G3 = *(const float4*)(f2p + 20);

        float p0  = (A0.x > 0) ? fmaxf(c1 * F0.x, c2 * G0.x) : 0.0f;
        float p1_ = (A0.y > 0) ? fmaxf(c1 * F0.y, c2 * G0.y) : 0.0f;
        float p2_ = (A0.z > 0) ? fmaxf(c1 * F0.z, c2 * G0.z) : 0.0f;
        float p3  = (A0.w > 0) ? fmaxf(c1 * F0.w, c2 * G0.w) : 0.0f;
        float p4  = (A1.x > 0) ? fmaxf(c1 * F1.x, c2 * G1.x) : 0.0f;
        float p5  = (A1.y > 0) ? fmaxf(c1 * F1.y, c2 * G1.y) : 0.0f;
        float p6  = (A1.z > 0) ? fmaxf(c1 * F1.z, c2 * G1.z) : 0.0f;
        float p7  = (A1.w > 0) ? fmaxf(c1 * F1.w, c2 * G1.w) : 0.0f;
        float p8  = (A2.x > 0) ? fmaxf(c1 * F2.x, c2 * G2.x) : 0.0f;
        float p9  = (A2.y > 0) ? fmaxf(c1 * F2.y, c2 * G2.y) : 0.0f;
        float p10 = (A2.z > 0) ? fmaxf(c1 * F2.z, c2 * G2.z) : 0.0f;
        float p11 = (A2.w > 0) ? fmaxf(c1 * F2.w, c2 * G2.w) : 0.0f;
        float p12 = (A3.x > 0) ? fmaxf(c1 * F3.x, c2 * G3.x) : 0.0f;
        float p13 = (A3.y > 0) ? fmaxf(c1 * F3.y, c2 * G3.y) : 0.0f;
        float p14 = (A3.z > 0) ? fmaxf(c1 * F3.z, c2 * G3.z) : 0.0f;
        float p15 = (A3.w > 0) ? fmaxf(c1 * F3.w, c2 * G3.w) : 0.0f;

        v8u au;
        au[0] = pk_bf16(p0, p1_);  au[1] = pk_bf16(p2_, p3);
        au[2] = pk_bf16(p4, p5);   au[3] = pk_bf16(p6, p7);
        au[4] = pk_bf16(p8, p9);   au[5] = pk_bf16(p10, p11);
        au[6] = pk_bf16(p12, p13); au[7] = pk_bf16(p14, p15);
        v16bf Av = __builtin_bit_cast(v16bf, au);

        const v8u* bp = (const v8u*)hB + (size_t)((k0 + jb) >> 5) * 128 + l;
        v16bf B0 = __builtin_bit_cast(v16bf, bp[0]);
        v16bf B1 = __builtin_bit_cast(v16bf, bp[32]);
        v16bf B2 = __builtin_bit_cast(v16bf, bp[64]);
        v16bf B3 = __builtin_bit_cast(v16bf, bp[96]);

        acc0 = __builtin_amdgcn_wmma_f32_16x16x32_bf16(false, Av, false, B0,    (short)0, acc0, false, false);
        acc1 = __builtin_amdgcn_wmma_f32_16x16x32_bf16(false, Av, false, B1,    (short)0, acc1, false, false);
        acc2 = __builtin_amdgcn_wmma_f32_16x16x32_bf16(false, Av, false, B2,    (short)0, acc2, false, false);
        acc3 = __builtin_amdgcn_wmma_f32_16x16x32_bf16(false, Av, false, B3,    (short)0, acc3, false, false);
        accS = __builtin_amdgcn_wmma_f32_16x16x32_bf16(false, Av, false, Bones, (short)0, accS, false, false);
    }

    // partial numerator: C layout row = m0 + r + 8*half, col = nt*16 + q
    float* ob = pnum + ((size_t)y * NN + m0 + 8 * half) * OF + q;
#pragma unroll
    for (int r = 0; r < 8; ++r) {
        ob[(size_t)r * OF + 0]  = acc0[r];
        ob[(size_t)r * OF + 16] = acc1[r];
        ob[(size_t)r * OF + 32] = acc2[r];
        ob[(size_t)r * OF + 48] = acc3[r];
    }
    // partial denominator: col 0 of the ones-product (rows r + 8*half)
    if (q == 0) {
        float* pso = ps + (size_t)y * NN + m0 + 8 * half;
#pragma unroll
        for (int r = 0; r < 8; ++r) pso[r] = accS[r];
    }
}

// ------------------------------------------------------------------
// Kernel 5: reduce split-K partials, normalize, ELU.
// ------------------------------------------------------------------
__global__ void __launch_bounds__(256) finalize_kernel(const float* __restrict__ pnum,
                                                       const float* __restrict__ ps,
                                                       float* __restrict__ out) {
    int gid = blockIdx.x * 256 + threadIdx.x;   // NN*OF total
    int i   = gid >> 6;
    float num = 0.0f, s = 0.0f;
#pragma unroll
    for (int y = 0; y < SPLITK; ++y) {
        num += pnum[(size_t)y * (NN * OF) + gid];
        s   += ps[(size_t)y * NN + i];
    }
    s = (s > 0.0f) ? s : 1.0f;       // guard: isolated node (prob ~0)
    float v = num / s;
    out[gid] = (v > 0.0f) ? v : (__expf(v) - 1.0f);
}

// ------------------------------------------------------------------
extern "C" void kernel_launch(void* const* d_in, const int* in_sizes, int n_in,
                              void* d_out, int out_size, void* d_ws, size_t ws_size,
                              hipStream_t stream) {
    const float* x   = (const float*)d_in[0];
    const int*   adj = (const int*)d_in[1];
    const float* W   = (const float*)d_in[2];
    const float* a   = (const float*)d_in[3];
    float*       out = (float*)d_out;

    char* ws = (char*)d_ws;
    unsigned* Wpk = (unsigned*)ws;                                  //  64 KB
    unsigned* hB  = (unsigned*)(ws + (64 << 10));                   //   1 MB
    float* e1   = (float*)(ws + (64 << 10) + (1 << 20));            //  32 KB
    float* e2   = e1 + NN;                                          //  32 KB
    float* g1   = e2 + NN;                                          //  32 KB
    float* g2   = g1 + NN;                                          //  32 KB
    float* gmax = g2 + NN;                                          //  pad 64
    float* ps   = gmax + 64;                                        // 256 KB
    float* pnum = ps + (size_t)SPLITK * NN;                         //  16 MB

    prep_w_kernel<<<dim3(64), dim3(256), 0, stream>>>(W, Wpk);
    gemm1_kernel<<<dim3(NN / 16 / 8), dim3(256), 0, stream>>>(x, Wpk, a, hB, e1, e2);
    e2prep_kernel<<<dim3(1), dim3(256), 0, stream>>>(e2, gmax, g1, g2);
    gemm2_kernel<<<dim3(NN / 16, SPLITK), dim3(32), 0, stream>>>(adj, hB, e1, g1, g2, gmax, pnum, ps);
    finalize_kernel<<<dim3(NN * OF / 256), dim3(256), 0, stream>>>(pnum, ps, out);
}
